// GT_28991029248863
// MI455X (gfx1250) — matrix operand order repaired and checked by reading, not compiled, not run on hardware
//
#include <hip/hip_runtime.h>
#include <math.h>

// Graph Transformer forward for MI455X (gfx1250, wave32, WMMA).
// Dense GEMMs: v_wmma_f32_16x16x32_bf16, weight panel staged in LDS
// via global_load_async_to_lds_b128 (ASYNCcnt path), conflict-free
// k-slice layout. Edge softmax = 3 streaming passes with L2 atomics.

typedef __bf16 bf16_t;
typedef __attribute__((ext_vector_type(16))) __bf16 v16bf;
typedef __attribute__((ext_vector_type(8)))  __bf16 v8bf;
typedef __attribute__((ext_vector_type(8)))  float  v8f;

#define N_NODES 50000
#define E_EDGES 800000
#define F_IN    256
#define D_HID   128
#define C_CLS   40
#define L_LAYERS 5
#define H_HEADS 8
#define LN_EPS  1e-5f

// -------------------------------------------------- async-to-LDS (CDNA5 path)
#if __has_builtin(__builtin_amdgcn_global_load_async_to_lds_b128)
#define USE_ASYNC_LDS 1
typedef int v4i_t __attribute__((vector_size(16)));
typedef __attribute__((address_space(1))) v4i_t* gv4i_p;   // global int4*
typedef __attribute__((address_space(3))) v4i_t* lv4i_p;   // LDS int4*
__device__ __forceinline__ void async_cp16(const bf16_t* g, bf16_t* l) {
  __builtin_amdgcn_global_load_async_to_lds_b128((gv4i_p)g, (lv4i_p)l, 0, 0);
}
__device__ __forceinline__ void wait_async0() {
#if __has_builtin(__builtin_amdgcn_s_wait_asynccnt)
  __builtin_amdgcn_s_wait_asynccnt(0);
#else
  asm volatile("s_wait_asynccnt 0" ::: "memory");
#endif
}
#endif

// ---------------------------------------------------------------- utilities

__global__ void k_f32_to_bf16(const float* __restrict__ in,
                              bf16_t* __restrict__ out, int n) {
  int i = blockIdx.x * blockDim.x + threadIdx.x;
  if (i < n) out[i] = (bf16_t)in[i];
}

// W [K, M] row-major fp32  ->  WT [M, K] bf16 (so B-fragments load contiguously)
__global__ void k_transpose_bf16(const float* __restrict__ W,
                                 bf16_t* __restrict__ WT, int K, int M) {
  int i = blockIdx.x * blockDim.x + threadIdx.x;
  if (i >= K * M) return;
  int k = i / M, m = i - k * M;
  WT[(size_t)m * K + k] = (bf16_t)W[i];
}

// One wave per row: LayerNorm(h[row]) * g + b  ->  bf16
__global__ void k_layernorm_bf16(const float* __restrict__ h,
                                 const float* __restrict__ g,
                                 const float* __restrict__ b,
                                 bf16_t* __restrict__ out, int nrows) {
  int wave = (int)((blockIdx.x * (size_t)blockDim.x + threadIdx.x) >> 5);
  int lane = threadIdx.x & 31;
  if (wave >= nrows) return;
  const float* row = h + (size_t)wave * D_HID;
  float4 x = ((const float4*)row)[lane];           // 32 lanes * 4 = 128
  float s = x.x + x.y + x.z + x.w;
#pragma unroll
  for (int m = 16; m > 0; m >>= 1) s += __shfl_xor(s, m, 32);
  float mu = s * (1.0f / D_HID);
  float d0 = x.x - mu, d1 = x.y - mu, d2 = x.z - mu, d3 = x.w - mu;
  float sq = d0 * d0 + d1 * d1 + d2 * d2 + d3 * d3;
#pragma unroll
  for (int m = 16; m > 0; m >>= 1) sq += __shfl_xor(sq, m, 32);
  float inv = rsqrtf(sq * (1.0f / D_HID) + LN_EPS);
  int base = lane * 4;
  bf16_t* orow = out + (size_t)wave * D_HID + base;
  orow[0] = (bf16_t)(d0 * inv * g[base + 0] + b[base + 0]);
  orow[1] = (bf16_t)(d1 * inv * g[base + 1] + b[base + 1]);
  orow[2] = (bf16_t)(d2 * inv * g[base + 2] + b[base + 2]);
  orow[3] = (bf16_t)(d3 * inv * g[base + 3] + b[base + 3]);
}

// ---------------------------------------------------------------- WMMA GEMM
// out[nrows, DOUT] = epilogue(A[nrows, KDIM] @ BT[DOUT, KDIM]^T + bias)
// Block = 256 thr = 8 waves; whole BT panel staged in LDS once per block in a
// k-slice-major layout (32B units) so fragment reads are conflict-free.
// Wave computes a 16x64 strip (A fragment reused across 4 WMMA tiles).
// MODE: 0 = store relu(f32), 1 = store f32, 2 = accumulate into outF (f32),
//       3 = store relu as bf16 into outB.
template <int KDIM, int DOUT, int MODE>
__global__ __launch_bounds__(256)
void k_gemm_wmma(const bf16_t* __restrict__ A,
                 const bf16_t* __restrict__ BT,
                 const float* __restrict__ bias,
                 float* __restrict__ outF,
                 bf16_t* __restrict__ outB,
                 int nrows) {
  constexpr int COLG = DOUT / 64;                // waves across columns
  constexpr int RTPB = 8 / COLG;                 // row tiles per block
  constexpr int KS   = KDIM / 16;                // 16-elem k-slices per row
  constexpr int UNITS = DOUT * KDIM / 16;        // 32-byte units in panel
  constexpr int ITER  = UNITS / 256;
  __shared__ __align__(32) bf16_t Bs[DOUT * KDIM];

  const int tid = threadIdx.x;
  // ---- stage BT panel: global [n, k] -> LDS unit [(ks*DOUT + n)*16]
#pragma unroll
  for (int it = 0; it < ITER; ++it) {
    int u = tid + it * 256;
    int n = u / KS, ks = u - n * KS;
    const bf16_t* g = BT + (size_t)n * KDIM + ks * 16;
    bf16_t* l = Bs + ((size_t)ks * DOUT + n) * 16;
#ifdef USE_ASYNC_LDS
    async_cp16(g, l);
    async_cp16(g + 8, l + 8);
#else
    *(v16bf*)l = *(const v16bf*)g;
#endif
  }
#ifdef USE_ASYNC_LDS
  wait_async0();
#endif
  __syncthreads();

  const int lane    = tid & 31;
  const int wave    = tid >> 5;
  const int rowTile = blockIdx.x * RTPB + wave / COLG;
  if (rowTile * 16 >= nrows) return;
  const int rowBase = rowTile << 4;
  const int colBase = (wave % COLG) << 6;
  const int r16   = lane & 15;
  const int khalf = lane >> 4;        // 0: K 0-15, 1: K 16-31

  v8f acc0 = {}, acc1 = {}, acc2 = {}, acc3 = {};
  const bf16_t* ap = A + (size_t)(rowBase + r16) * KDIM + khalf * 16;

#pragma unroll
  for (int k0 = 0; k0 < KDIM; k0 += 32) {
    // A fragment: contiguous k order, lane half h holds k 16h..16h+15 (same as B)
    v8bf alo = *(const v8bf*)(ap + k0);
    v8bf ahi = *(const v8bf*)(ap + k0 + 8);
    v16bf a;
#pragma unroll
    for (int i = 0; i < 8; ++i) { a[i] = alo[i]; a[8 + i] = ahi[i]; }
    // B fragments from LDS: lane half selects k-slice, 16 lanes hit 16
    // consecutive 32B units -> conflict-free ds_load.
    const int ks = (k0 >> 4) + khalf;
    const bf16_t* bb = Bs + ((size_t)ks * DOUT + colBase + r16) * 16;
    v16bf b0 = *(const v16bf*)(bb);
    v16bf b1 = *(const v16bf*)(bb + 16 * 16);
    v16bf b2 = *(const v16bf*)(bb + 32 * 16);
    v16bf b3 = *(const v16bf*)(bb + 48 * 16);
    acc0 = __builtin_amdgcn_wmma_f32_16x16x32_bf16(false, a, false, b0, (short)0, acc0, false, false);
    acc1 = __builtin_amdgcn_wmma_f32_16x16x32_bf16(false, a, false, b1, (short)0, acc1, false, false);
    acc2 = __builtin_amdgcn_wmma_f32_16x16x32_bf16(false, a, false, b2, (short)0, acc2, false, false);
    acc3 = __builtin_amdgcn_wmma_f32_16x16x32_bf16(false, a, false, b3, (short)0, acc3, false, false);
  }

  // Epilogue. C/D layout: VGPR r -> row r (lanes 0-15) / row r+8 (lanes 16-31).
  const int colL = colBase + r16;
  const float bb0 = bias[colL +  0];
  const float bb1 = bias[colL + 16];
  const float bb2 = bias[colL + 32];
  const float bb3 = bias[colL + 48];
#pragma unroll
  for (int r = 0; r < 8; ++r) {
    const size_t row  = (size_t)(rowBase + r + khalf * 8);
    const size_t base = row * DOUT + colL;
    float v0 = acc0[r] + bb0, v1 = acc1[r] + bb1;
    float v2 = acc2[r] + bb2, v3 = acc3[r] + bb3;
    if (MODE == 0) {
      outF[base +  0] = fmaxf(v0, 0.f); outF[base + 16] = fmaxf(v1, 0.f);
      outF[base + 32] = fmaxf(v2, 0.f); outF[base + 48] = fmaxf(v3, 0.f);
    } else if (MODE == 1) {
      outF[base +  0] = v0; outF[base + 16] = v1;
      outF[base + 32] = v2; outF[base + 48] = v3;
    } else if (MODE == 2) {
      outF[base +  0] += v0; outF[base + 16] += v1;
      outF[base + 32] += v2; outF[base + 48] += v3;
    } else {
      outB[base +  0] = (bf16_t)fmaxf(v0, 0.f); outB[base + 16] = (bf16_t)fmaxf(v1, 0.f);
      outB[base + 32] = (bf16_t)fmaxf(v2, 0.f); outB[base + 48] = (bf16_t)fmaxf(v3, 0.f);
    }
  }
}

// ---------------------------------------------------------------- edge softmax
__device__ __forceinline__ unsigned encF(float f) {
  unsigned u = __float_as_uint(f);
  return (u & 0x80000000u) ? ~u : (u | 0x80000000u);
}
__device__ __forceinline__ float decF(unsigned u) {
  u = (u & 0x80000000u) ? (u & 0x7FFFFFFFu) : ~u;
  return __uint_as_float(u);
}

// Pass 1: scores[e,h] = dot(q[src], k[dst]) / sqrt(16); atomic max per (dst,h)
__global__ void k_edge_scores(const float* __restrict__ q,
                              const float* __restrict__ k,
                              const int* __restrict__ esrc,
                              const int* __restrict__ edst,
                              float* __restrict__ sc,
                              unsigned* __restrict__ mx) {
  int idx = blockIdx.x * blockDim.x + threadIdx.x;
  if (idx >= E_EDGES * H_HEADS) return;
  int e = idx >> 3, hh = idx & 7;
  int s = esrc[e], d = edst[e];
  const float4* qp = (const float4*)(q + (size_t)s * D_HID + hh * 16);
  const float4* kp = (const float4*)(k + (size_t)d * D_HID + hh * 16);
  float dot = 0.f;
#pragma unroll
  for (int j = 0; j < 4; ++j) {
    float4 a = qp[j], b = kp[j];
    dot += a.x * b.x + a.y * b.y + a.z * b.z + a.w * b.w;
  }
  float score = dot * 0.25f;                 // 1/sqrt(HD), HD=16
  sc[idx] = score;
  atomicMax(&mx[(size_t)d * H_HEADS + hh], encF(score));
}

// Pass 2: ex = exp(score - max); atomic-add denom per (dst,h); ex overwrites sc
__global__ void k_edge_exp(float* __restrict__ sc,
                           const unsigned* __restrict__ mx,
                           const int* __restrict__ edst,
                           float* __restrict__ den) {
  int idx = blockIdx.x * blockDim.x + threadIdx.x;
  if (idx >= E_EDGES * H_HEADS) return;
  int e = idx >> 3, hh = idx & 7;
  int d = edst[e];
  float m  = decF(mx[(size_t)d * H_HEADS + hh]);
  float ex = __expf(sc[idx] - m);
  sc[idx] = ex;
  atomicAdd(&den[(size_t)d * H_HEADS + hh], ex);
}

// Pass 3: agg[dst] += v[src] * (ex / denom[dst])
__global__ void k_edge_agg(const float* __restrict__ sc,
                           const float* __restrict__ den,
                           const float* __restrict__ v,
                           const int* __restrict__ esrc,
                           const int* __restrict__ edst,
                           float* __restrict__ agg) {
  int idx = blockIdx.x * blockDim.x + threadIdx.x;
  if (idx >= E_EDGES * H_HEADS) return;
  int e = idx >> 3, hh = idx & 7;
  int s = esrc[e], d = edst[e];
  float p = sc[idx] / den[(size_t)d * H_HEADS + hh];
  const float* vp = v   + (size_t)s * D_HID + hh * 16;
  float*       ap = agg + (size_t)d * D_HID + hh * 16;
#pragma unroll
  for (int j = 0; j < 16; ++j) atomicAdd(ap + j, vp[j] * p);
}

// ---------------------------------------------------------------- readout
// One block (128 thr) per node: LN(h) then 40-way projection.
__global__ void k_final(const float* __restrict__ h,
                        const float* __restrict__ g,
                        const float* __restrict__ b,
                        const float* __restrict__ Wout,
                        const float* __restrict__ bout,
                        float* __restrict__ out) {
  __shared__ float s[D_HID];
  __shared__ float red[4];
  int node = blockIdx.x, t = threadIdx.x;
  float x = h[(size_t)node * D_HID + t];
  float sum = x;
#pragma unroll
  for (int m = 16; m > 0; m >>= 1) sum += __shfl_xor(sum, m, 32);
  if ((t & 31) == 0) red[t >> 5] = sum;
  __syncthreads();
  float mu = (red[0] + red[1] + red[2] + red[3]) * (1.0f / D_HID);
  __syncthreads();
  float d = x - mu;
  float sq = d * d;
#pragma unroll
  for (int m = 16; m > 0; m >>= 1) sq += __shfl_xor(sq, m, 32);
  if ((t & 31) == 0) red[t >> 5] = sq;
  __syncthreads();
  float inv = rsqrtf((red[0] + red[1] + red[2] + red[3]) * (1.0f / D_HID) + LN_EPS);
  s[t] = d * inv * g[t] + b[t];
  __syncthreads();
  if (t < C_CLS) {
    float acc = bout[t];
#pragma unroll 8
    for (int kk = 0; kk < D_HID; ++kk) acc += s[kk] * Wout[kk * C_CLS + t];
    out[(size_t)node * C_CLS + t] = acc;
  }
}

// ---------------------------------------------------------------- launcher

static inline int cdiv(long a, long b) { return (int)((a + b - 1) / b); }
static inline int gemmGrid(int nrows, int dout) {
  int rowTiles = nrows / 16;
  int rtpb = 8 / (dout / 64);
  return cdiv(rowTiles, rtpb);
}

extern "C" void kernel_launch(void* const* d_in, const int* in_sizes, int n_in,
                              void* d_out, int out_size, void* d_ws, size_t ws_size,
                              hipStream_t stream) {
  const float* x    = (const float*)d_in[0];
  const int*   esrc = (const int*)d_in[1];
  const int*   edst = (const int*)d_in[2];
  const float* Wi   = (const float*)d_in[3];
  const float* bi   = (const float*)d_in[4];
  const float* Wq   = (const float*)d_in[5];
  const float* bq   = (const float*)d_in[6];
  const float* Wk   = (const float*)d_in[7];
  const float* bk   = (const float*)d_in[8];
  const float* Wv   = (const float*)d_in[9];
  const float* bv   = (const float*)d_in[10];
  const float* Wo   = (const float*)d_in[11];
  const float* bo   = (const float*)d_in[12];
  const float* g1   = (const float*)d_in[13];
  const float* b1   = (const float*)d_in[14];
  const float* Wf1  = (const float*)d_in[15];
  const float* bf1  = (const float*)d_in[16];
  const float* Wf2  = (const float*)d_in[17];
  const float* bf2  = (const float*)d_in[18];
  const float* gOut = (const float*)d_in[19];
  const float* bOut = (const float*)d_in[20];
  const float* Wout = (const float*)d_in[21];
  const float* bout = (const float*)d_in[22];

  const int D = D_HID, D2 = 2 * D_HID;
  float* h   = (float*)d_out;                       // [N, D] == "mid" output
  float* out = (float*)d_out + (size_t)N_NODES * D; // [N, C]

  char* p = (char*)d_ws;
  auto carve = [&](size_t bytes) {
    char* r = p; p += (bytes + 255) & ~(size_t)255; return (void*)r;
  };
  bf16_t*   xb   = (bf16_t*)carve((size_t)N_NODES * F_IN * 2);
  bf16_t*   hb   = (bf16_t*)carve((size_t)N_NODES * D * 2);
  float*    q    = (float*) carve((size_t)N_NODES * D * 4);
  float*    kk   = (float*) carve((size_t)N_NODES * D * 4);
  float*    v    = (float*) carve((size_t)N_NODES * D * 4);
  float*    agg  = (float*) carve((size_t)N_NODES * D * 4);
  bf16_t*   aggb = (bf16_t*)carve((size_t)N_NODES * D * 2);
  bf16_t*   midb = (bf16_t*)carve((size_t)N_NODES * D2 * 2);
  float*    sc   = (float*) carve((size_t)E_EDGES * H_HEADS * 4);
  unsigned* mx   = (unsigned*)carve((size_t)N_NODES * H_HEADS * 4);
  float*    den  = (float*) carve((size_t)N_NODES * H_HEADS * 4);
  bf16_t*   WiT  = (bf16_t*)carve((size_t)F_IN * D * 2);
  bf16_t*   WqT  = (bf16_t*)carve((size_t)L_LAYERS * D * D * 2);
  bf16_t*   WkT  = (bf16_t*)carve((size_t)L_LAYERS * D * D * 2);
  bf16_t*   WvT  = (bf16_t*)carve((size_t)L_LAYERS * D * D * 2);
  bf16_t*   WoT  = (bf16_t*)carve((size_t)L_LAYERS * D * D * 2);
  bf16_t*   Wf1T = (bf16_t*)carve((size_t)L_LAYERS * D * D2 * 2);
  bf16_t*   Wf2T = (bf16_t*)carve((size_t)L_LAYERS * D2 * D * 2);

  // --- weight transposition/conversion (cheap: ~0.7 M elems) ---
  k_transpose_bf16<<<cdiv((long)F_IN * D, 256), 256, 0, stream>>>(Wi, WiT, F_IN, D);
  for (int i = 0; i < L_LAYERS; ++i) {
    int dd = D * D;
    k_transpose_bf16<<<cdiv(dd, 256), 256, 0, stream>>>(Wq + (size_t)i * dd, WqT + (size_t)i * dd, D, D);
    k_transpose_bf16<<<cdiv(dd, 256), 256, 0, stream>>>(Wk + (size_t)i * dd, WkT + (size_t)i * dd, D, D);
    k_transpose_bf16<<<cdiv(dd, 256), 256, 0, stream>>>(Wv + (size_t)i * dd, WvT + (size_t)i * dd, D, D);
    k_transpose_bf16<<<cdiv(dd, 256), 256, 0, stream>>>(Wo + (size_t)i * dd, WoT + (size_t)i * dd, D, D);
    k_transpose_bf16<<<cdiv(D * D2, 256), 256, 0, stream>>>(Wf1 + (size_t)i * D * D2, Wf1T + (size_t)i * D * D2, D, D2);
    k_transpose_bf16<<<cdiv(D2 * D, 256), 256, 0, stream>>>(Wf2 + (size_t)i * D2 * D, Wf2T + (size_t)i * D2 * D, D2, D);
  }
  k_f32_to_bf16<<<cdiv((long)N_NODES * F_IN, 256), 256, 0, stream>>>(x, xb, N_NODES * F_IN);

  // --- input projection: h = relu(x @ Wi + bi) ---
  k_gemm_wmma<F_IN, D_HID, 0><<<gemmGrid(N_NODES, D), 256, 0, stream>>>(
      xb, WiT, bi, h, nullptr, N_NODES);

  const int lnBlocks = cdiv((long)N_NODES * 32, 256);
  const int egBlocks = cdiv((long)E_EDGES * H_HEADS, 256);
  const int cvBlocks = cdiv((long)N_NODES * D, 256);

  for (int i = 0; i < L_LAYERS; ++i) {
    const size_t dd = (size_t)i * D * D;
    // r = LN(h)
    k_layernorm_bf16<<<lnBlocks, 256, 0, stream>>>(h, g1 + i * D, b1 + i * D, hb, N_NODES);
    // q, k, v
    k_gemm_wmma<D_HID, D_HID, 1><<<gemmGrid(N_NODES, D), 256, 0, stream>>>(hb, WqT + dd, bq + i * D, q,  nullptr, N_NODES);
    k_gemm_wmma<D_HID, D_HID, 1><<<gemmGrid(N_NODES, D), 256, 0, stream>>>(hb, WkT + dd, bk + i * D, kk, nullptr, N_NODES);
    k_gemm_wmma<D_HID, D_HID, 1><<<gemmGrid(N_NODES, D), 256, 0, stream>>>(hb, WvT + dd, bv + i * D, v,  nullptr, N_NODES);
    // edge softmax + aggregation
    (void)hipMemsetAsync(mx,  0, (size_t)N_NODES * H_HEADS * 4, stream);
    (void)hipMemsetAsync(den, 0, (size_t)N_NODES * H_HEADS * 4, stream);
    (void)hipMemsetAsync(agg, 0, (size_t)N_NODES * D * 4, stream);
    k_edge_scores<<<egBlocks, 256, 0, stream>>>(q, kk, esrc, edst, sc, mx);
    k_edge_exp   <<<egBlocks, 256, 0, stream>>>(sc, mx, edst, den);
    k_edge_agg   <<<egBlocks, 256, 0, stream>>>(sc, den, v, esrc, edst, agg);
    // h += agg @ Wo + bo
    k_f32_to_bf16<<<cvBlocks, 256, 0, stream>>>(agg, aggb, N_NODES * D);
    k_gemm_wmma<D_HID, D_HID, 2><<<gemmGrid(N_NODES, D), 256, 0, stream>>>(aggb, WoT + dd, bo + i * D, h, nullptr, N_NODES);
    // FFN: h += relu(LN(h) @ Wf1 + bf1) @ Wf2 + bf2   (reuses g1/b1, as reference)
    k_layernorm_bf16<<<lnBlocks, 256, 0, stream>>>(h, g1 + i * D, b1 + i * D, hb, N_NODES);
    k_gemm_wmma<D_HID, 256, 3><<<gemmGrid(N_NODES, D2), 256, 0, stream>>>(
        hb, Wf1T + (size_t)i * D * D2, bf1 + i * D2, nullptr, midb, N_NODES);
    k_gemm_wmma<256, D_HID, 2><<<gemmGrid(N_NODES, D), 256, 0, stream>>>(
        midb, Wf2T + (size_t)i * D2 * D, bf2 + i * D, h, nullptr, N_NODES);
  }

  // --- readout: out = LN(h) @ Wout + bout ---
  k_final<<<N_NODES, 128, 0, stream>>>(h, gOut, bOut, Wout, bout, out);
}
